// GCN_54614804136132
// MI455X (gfx1250) — matrix operand order repaired and checked by reading, not compiled, not run on hardware
//
#include <hip/hip_runtime.h>
#include <hip/hip_bf16.h>
#include <math.h>
#include <stdint.h>

typedef __attribute__((ext_vector_type(16))) _Float16 v16h;
typedef __attribute__((ext_vector_type(8)))  float    v8f;
typedef __attribute__((ext_vector_type(4)))  float    v4f;

#define F_IN 256
#define HID  128

__device__ __forceinline__ void atomAddF(float* p, float v) {
    (void)__hip_atomic_fetch_add(p, v, __ATOMIC_RELAXED, __HIP_MEMORY_SCOPE_AGENT);
}

// ---- degree / normalization -------------------------------------------------
__global__ __launch_bounds__(256) void k_deg_init(float* deg, int N) {
    int i = blockIdx.x * blockDim.x + threadIdx.x;
    if (i < N) deg[i] = 1.0f;   // self-loop
}

__global__ __launch_bounds__(256) void k_deg_scatter(const int* __restrict__ dst,
                                                     float* deg, int E) {
    int e = blockIdx.x * blockDim.x + threadIdx.x;
    if (e < E) atomAddF(&deg[dst[e]], 1.0f);
}

__global__ __launch_bounds__(256) void k_dinv(float* deg, int N) {
    int i = blockIdx.x * blockDim.x + threadIdx.x;
    if (i < N) deg[i] = rsqrtf(deg[i]);   // in-place: deg -> dinv
}

// ---- pre-swizzle W1 (f32 [256,128] row-major) into f16 B fragments ----------
// Fragment fi = (c*8 + t)*32 + lane holds 16 contiguous halves:
//   n = 16*c + (lane&15), k = 32*t + 8*(lane>>4) + (j&7) + 16*(j>>3)  (j = 0..15)
__global__ __launch_bounds__(256) void k_prep_w1(const float* __restrict__ W1,
                                                 _Float16* __restrict__ W1s) {
    int tid = blockIdx.x * blockDim.x + threadIdx.x;   // 0 .. 32767
    int j  = tid & 15;
    int fi = tid >> 4;
    int l  = fi & 31;
    int t  = (fi >> 5) & 7;
    int c  = fi >> 8;
    int n  = 16 * c + (l & 15);
    int k  = 32 * t + ((l >> 4) << 3) + (j & 7) + ((j >> 3) << 4);
    W1s[tid] = (_Float16)W1[k * HID + n];
}

// ---- layer-1 GEMM: y1 = (x @ W1) * dinv[row]; agg1 initialized = y1 ---------
// block = 16-row tile (8 waves), wave w = column tile n0 = 16*w.
__global__ __launch_bounds__(256)
void k_gemm1(const float* __restrict__ x, const _Float16* __restrict__ W1s,
             const float* __restrict__ dinv, float* __restrict__ y1,
             float* __restrict__ agg1, int N) {
    const int lane = threadIdx.x & 31;
    const int w    = threadIdx.x >> 5;       // column tile 0..7
    const int r0   = blockIdx.x * 16;
    const int hi   = lane >> 4;              // 0 or 1
    int arow = r0 + (lane & 15);
    if (arow >= N) arow = N - 1;             // branchless clamp keeps EXEC all-1s
    const float* xr = x + (size_t)arow * F_IN;
    const int kbA = hi * 8;                  // A K-split per lane half

    const _Float16* bbase = W1s + (size_t)w * 4096 + (size_t)lane * 16;

    v8f acc = {};
    #pragma unroll
    for (int t = 0; t < 8; ++t) {
        const int k0 = t * 32;
        v4f a0 = *(const v4f*)(xr + k0 + kbA);
        v4f a1 = *(const v4f*)(xr + k0 + kbA + 4);
        v4f a2 = *(const v4f*)(xr + k0 + kbA + 16);
        v4f a3 = *(const v4f*)(xr + k0 + kbA + 20);
        v16h a;
        #pragma unroll
        for (int j = 0; j < 4; ++j) {
            a[j]      = (_Float16)a0[j];
            a[4 + j]  = (_Float16)a1[j];
            a[8 + j]  = (_Float16)a2[j];
            a[12 + j] = (_Float16)a3[j];
        }
        const v16h b = *(const v16h*)(bbase + (size_t)t * 512);
        acc = __builtin_amdgcn_wmma_f32_16x16x32_f16(
                  false, a, false, b, (short)0, acc, false, false);
    }

    // epilogue: C/D layout -> lane holds N=lane&15, rows mb..mb+7 (mb = 8*hi)
    const int n  = w * 16 + (lane & 15);
    const int mb = hi * 8;
    #pragma unroll
    for (int r = 0; r < 8; ++r) {
        int row = r0 + mb + r;
        if (row < N) {
            float v = acc[r] * dinv[row];
            y1[(size_t)row * HID + n]   = v;
            agg1[(size_t)row * HID + n] = v;   // self-loop term
        }
    }
}

// ---- layer-1 aggregation: wave per edge, 4 channels per lane ----------------
__global__ __launch_bounds__(256)
void k_agg1(const int* __restrict__ src, const int* __restrict__ dst,
            const float* __restrict__ y1, float* __restrict__ agg1, int E) {
    int gid  = blockIdx.x * blockDim.x + threadIdx.x;
    int e    = gid >> 5;
    int lane = gid & 31;
    if (e >= E) return;
    int s = src[e], d = dst[e];
    v4f v = *(const v4f*)(y1 + (size_t)s * HID + lane * 4);
    float* ad = agg1 + (size_t)d * HID + lane * 4;
    atomAddF(ad + 0, v[0]);
    atomAddF(ad + 1, v[1]);
    atomAddF(ad + 2, v[2]);
    atomAddF(ad + 3, v[3]);
}

// ---- h = relu(dinv*agg1 + b1); y2 = (h @ W2) * dinv; agg2 = y2 --------------
// wave per node; 4 channels per lane; wave32 shuffle reduction.
__global__ __launch_bounds__(256)
void k_layer2(const float* __restrict__ agg1, const float* __restrict__ dinv,
              const float* __restrict__ b1, const float* __restrict__ W2,
              float* __restrict__ y2, float* __restrict__ agg2, int N) {
    int gid  = blockIdx.x * blockDim.x + threadIdx.x;
    int i    = gid >> 5;
    int lane = gid & 31;
    if (i >= N) return;
    float di = dinv[i];
    float o0 = 0.f, o1 = 0.f;
    #pragma unroll
    for (int s = 0; s < 4; ++s) {
        int ch = lane + 32 * s;
        float hc = fmaxf(di * agg1[(size_t)i * HID + ch] + b1[ch], 0.0f);
        o0 += hc * W2[ch * 2 + 0];
        o1 += hc * W2[ch * 2 + 1];
    }
    #pragma unroll
    for (int off = 16; off >= 1; off >>= 1) {
        o0 += __shfl_xor(o0, off, 32);
        o1 += __shfl_xor(o1, off, 32);
    }
    if (lane == 0) {
        float v0 = di * o0, v1 = di * o1;
        y2[2 * i]     = v0;  y2[2 * i + 1]   = v1;
        agg2[2 * i]   = v0;  agg2[2 * i + 1] = v1;   // self-loop term
    }
}

// ---- layer-2 aggregation: thread per edge (2 channels) ----------------------
__global__ __launch_bounds__(256)
void k_agg2(const int* __restrict__ src, const int* __restrict__ dst,
            const float* __restrict__ y2, float* __restrict__ agg2, int E) {
    int e = blockIdx.x * blockDim.x + threadIdx.x;
    if (e >= E) return;
    int s = src[e], d = dst[e];
    atomAddF(&agg2[2 * d],     y2[2 * s]);
    atomAddF(&agg2[2 * d + 1], y2[2 * s + 1]);
}

// ---- final scale + bias + softmax over 2 classes ----------------------------
__global__ __launch_bounds__(256)
void k_softmax(const float* __restrict__ agg2, const float* __restrict__ dinv,
               const float* __restrict__ b2, float* __restrict__ out, int N) {
    int i = blockIdx.x * blockDim.x + threadIdx.x;
    if (i >= N) return;
    float di = dinv[i];
    float o0 = di * agg2[2 * i]     + b2[0];
    float o1 = di * agg2[2 * i + 1] + b2[1];
    float m  = fmaxf(o0, o1);
    float e0 = __expf(o0 - m), e1 = __expf(o1 - m);
    float inv = 1.0f / (e0 + e1);
    out[2 * i]     = e0 * inv;
    out[2 * i + 1] = e1 * inv;
}

extern "C" void kernel_launch(void* const* d_in, const int* in_sizes, int n_in,
                              void* d_out, int out_size, void* d_ws, size_t ws_size,
                              hipStream_t stream) {
    const float* x   = (const float*)d_in[0];
    const int*   ei  = (const int*)d_in[1];
    const float* W1  = (const float*)d_in[2];
    const float* b1  = (const float*)d_in[3];
    const float* W2  = (const float*)d_in[4];
    const float* b2  = (const float*)d_in[5];
    float*       out = (float*)d_out;

    const int N = in_sizes[0] / F_IN;
    const int E = in_sizes[1] / 2;
    const int* srcp = ei;
    const int* dstp = ei + E;

    // workspace carve-up (float units), W1s aligned up to 32B
    float* dinv = (float*)d_ws;                     // N
    float* y1   = dinv + N;                         // N*HID
    float* agg1 = y1 + (size_t)N * HID;             // N*HID
    float* y2   = agg1 + (size_t)N * HID;           // 2N
    float* agg2 = y2 + 2 * (size_t)N;               // 2N
    uintptr_t wp = (uintptr_t)(agg2 + 2 * (size_t)N);
    wp = (wp + 31u) & ~(uintptr_t)31u;
    _Float16* W1s = (_Float16*)wp;                  // 32768 halves (64KB)

    const int T = 256;
    k_deg_init   <<<(N + T - 1) / T, T, 0, stream>>>(dinv, N);
    k_deg_scatter<<<(E + T - 1) / T, T, 0, stream>>>(dstp, dinv, E);
    k_dinv       <<<(N + T - 1) / T, T, 0, stream>>>(dinv, N);
    k_prep_w1    <<<32768 / T, T, 0, stream>>>(W1, W1s);

    k_gemm1<<<(N + 15) / 16, 256, 0, stream>>>(x, W1s, dinv, y1, agg1, N);

    long long aggThreads = (long long)E * 32;
    k_agg1<<<(int)((aggThreads + T - 1) / T), T, 0, stream>>>(srcp, dstp, y1, agg1, E);

    long long l2Threads = (long long)N * 32;
    k_layer2<<<(int)((l2Threads + T - 1) / T), T, 0, stream>>>(agg1, dinv, b1, W2, y2, agg2, N);

    k_agg2   <<<(E + T - 1) / T, T, 0, stream>>>(srcp, dstp, y2, agg2, E);
    k_softmax<<<(N + T - 1) / T, T, 0, stream>>>(agg2, dinv, b2, out, N);
}
